// MultiScaleMHAttentionMap_12515534701208
// MI455X (gfx1250) — matrix operand order repaired and checked by reading, not compiled, not run on hardware
//
#include <hip/hip_runtime.h>
#include <hip/hip_bf16.h>

// ---------------------------------------------------------------------------
// MultiScaleMHAttentionMap for MI455X (gfx1250), wave32 + WMMA bf16.
// B=2, Q=300, QD=256, HID=256, NH=8 (head dim 32), NL=4,
// P = {128*128, 64*64, 32*32, 16*16}.
//
// Roofline: ~12 GFLOP vs ~1.7 GB of traffic (418 MB scores written, softmax
// re-read/write) -> bandwidth bound at 23.3 TB/s. bf16 WMMA (f32 accum) for
// all GEMMs; K=32 of v_wmma_f32_16x16x32_bf16 == head dim exactly.
// ---------------------------------------------------------------------------

#define Bc    2
#define Qc    300
#define QDc   256
#define HIDc  256
#define NHc   8
#define HDc   32                   // head dim = HID/NH
#define NLc   4
#define QT    19                   // ceil(300/16)
#define HT    16                   // 256/16
#define KSTEPS 8                   // 256 / 32
#define MT    4                    // h-tiles per proj_k wave (share B fragment)
#define NPT   4                    // p-tiles per attn wave (share A fragment)
#define NORMc 0.17677669529663687f // 32^-0.5

typedef __attribute__((ext_vector_type(16))) __bf16 v16bf;
typedef __attribute__((ext_vector_type(8)))  float  v8f;

// ---------------------------------------------------------------------------
// Fragment layouts (gfx1250 wave32, CDNA5 ISA):
//  A (16x32 bf16): lane l -> row M = l&15; K(j) = 16*(l>>4) + j (contiguous, as loaded here)
//  B (32x16 bf16): lane l -> col N = l&15; K(j) = 16*(l>>4) + j (order used here, same as A; not the ISA table order)
//  C/D (16x16 f32): lane l -> col N = l&15; vgpr r -> row M = r + 8*(l>>4)
//
// kproj is stored B-fragment-swizzled: tile t = ((b*NH+head)*PT + pt),
// element [t*512 + lane*16 + j] = k_lvl[b][head*32 + 16*(lane>>4) + j][pt*16 + (lane&15)]
// so an attn B fragment is 32 contiguous bytes per lane.
// ---------------------------------------------------------------------------

// Q projection: qproj[l][b][q][h] = bf16( (q[b,q,:]·w_q[l][h,:] + b_q[l][h]) * NORM )
__global__ void __launch_bounds__(256)
proj_q_kernel(const float* __restrict__ q,    // [B,Q,QD]
              const float* __restrict__ w_q,  // [NL,HID,QD]
              const float* __restrict__ b_q,  // [NL,HID]
              __bf16* __restrict__ qproj) {   // [NL,B,Q,HID]
  int wid  = (blockIdx.x * blockDim.x + threadIdx.x) >> 5;
  int lane = threadIdx.x & 31;
  const int total = NLc * Bc * QT * HT;
  if (wid >= total) return;
  int ht = wid % HT;  int tmp = wid / HT;
  int qt = tmp % QT;  tmp /= QT;
  int b  = tmp % Bc;  int l = tmp / Bc;
  int q0 = qt * 16, h0 = ht * 16;

  // Clamp out-of-range rows instead of predicating loads: rows >= Q of A only
  // feed rows >= Q of D, which are never stored.
  int mrow   = q0 + (lane & 15);
  int mclamp = mrow < Qc ? mrow : (Qc - 1);
  int khalf  = (lane >> 4) * 16;
  const float* arow = q   + ((size_t)b * Qc + mclamp) * QDc + khalf;
  const float* brow = w_q + ((size_t)l * HIDc + h0 + (lane & 15)) * QDc
                          + ((lane >> 4) << 4);

  v8f c = {};
#pragma unroll
  for (int kk = 0; kk < KSTEPS; ++kk) {
    v16bf a, bm;
    const float* ap = arow + kk * 32;
#pragma unroll
    for (int j = 0; j < 16; ++j)
      a[j] = (__bf16)ap[j];   // 16 contiguous floats
    const float* bp = brow + kk * 32;
#pragma unroll
    for (int j = 0; j < 16; ++j) bm[j] = (__bf16)bp[j]; // 16 contiguous floats
    c = __builtin_amdgcn_wmma_f32_16x16x32_bf16(false, a, false, bm,
                                                (short)0, c, false, false);
  }

  int   n    = h0 + (lane & 15);
  float bias = b_q[l * HIDc + n];
#pragma unroll
  for (int r = 0; r < 8; ++r) {
    int m = q0 + r + ((lane >> 4) << 3);
    if (m < Qc)
      qproj[(((size_t)l * Bc + b) * Qc + m) * HIDc + n] =
          (__bf16)((c[r] + bias) * NORMc);
  }
}

// K projection (1x1 conv), output in B-fragment-swizzled layout.
// Each wave computes MT h-tiles (64 output channels) so the strided-load B
// fragment is shared by MT WMMAs per k-step.
__global__ void __launch_bounds__(256)
proj_k_kernel(const float* __restrict__ kin,   // [B,QD,P]
              const float* __restrict__ w_k,   // [NL,HID,QD]
              const float* __restrict__ b_k,   // [NL,HID]
              __bf16* __restrict__ kproj,      // swizzled, this level
              int l, int P) {
  int wid  = (blockIdx.x * blockDim.x + threadIdx.x) >> 5;
  int lane = threadIdx.x & 31;
  int PT = P >> 4;
  const int HTG = HT / MT;
  int total = Bc * HTG * PT;
  if (wid >= total) return;
  int pt  = wid % PT;   int tmp = wid / PT;
  int htg = tmp % HTG;  int b = tmp / HTG;
  int h0 = htg * (16 * MT), p0 = pt * 16;

  int khalf = (lane >> 4) * 16;
  const float* arow  = w_k + ((size_t)l * HIDc + h0 + (lane & 15)) * QDc + khalf;
  const float* bbase = kin + (size_t)b * QDc * P + (p0 + (lane & 15))
                           + (size_t)((lane >> 4) << 4) * P;

  v8f c[MT];
#pragma unroll
  for (int i = 0; i < MT; ++i) c[i] = (v8f){};

#pragma unroll
  for (int kk = 0; kk < KSTEPS; ++kk) {
    v16bf bm;
    const float* bp = bbase + (size_t)(kk * 32) * P;
#pragma unroll
    for (int j = 0; j < 16; ++j) bm[j] = (__bf16)bp[(size_t)j * P]; // 64B/lane-group
#pragma unroll
    for (int i = 0; i < MT; ++i) {
      v16bf a;
      const float* ap = arow + (size_t)i * 16 * QDc + kk * 32;
#pragma unroll
      for (int j = 0; j < 16; ++j)
        a[j] = (__bf16)ap[j];
      c[i] = __builtin_amdgcn_wmma_f32_16x16x32_bf16(false, a, false, bm,
                                                     (short)0, c[i], false, false);
    }
  }

  // Swizzled store: D row r (h = h0i + r + 8*(lane>>4)) maps to element
  // j = r + 8*(lane>>4) of fragment-lane l' = 16*half + (lane&15).
  // The 8 rows per lane are 8 consecutive bf16 -> one 16-byte store each.
#pragma unroll
  for (int i = 0; i < MT; ++i) {
    int h0i  = h0 + i * 16;
    int head = h0i >> 5;
    int half = (h0i >> 4) & 1;
    size_t tbase = ((size_t)(b * NHc + head) * PT + pt) * 512;
    union { __bf16 h[8]; uint4 v; } pack;
#pragma unroll
    for (int r = 0; r < 8; ++r) {
      int h = h0i + r + ((lane >> 4) << 3);
      pack.h[r] = (__bf16)(c[i][r] + b_k[l * HIDc + h]);
    }
    size_t off = tbase + (size_t)((half << 4) + (lane & 15)) * 16 + ((lane >> 4) << 3);
    *(uint4*)(kproj + off) = pack.v;
  }
}

// Attention scores: NPT p-tiles per wave, one wmma each (K=32 = head dim),
// masked, raw scores to d_out in final [B,Q,NH,P] layout.
__global__ void __launch_bounds__(256)
attn_kernel(const __bf16* __restrict__ qproj,       // [NL,B,Q,HID]
            const __bf16* __restrict__ kproj,       // swizzled, this level
            const unsigned char* __restrict__ mask, // [B,P] bool
            float* __restrict__ out,                // [B,Q,NH,P] (this level)
            int l, int P) {
  int wid  = (blockIdx.x * blockDim.x + threadIdx.x) >> 5;
  int lane = threadIdx.x & 31;
  int PT  = P >> 4;
  int PT4 = PT / NPT;
  int total = Bc * NHc * QT * PT4;
  if (wid >= total) return;
  int pt4 = wid % PT4;  int tmp = wid / PT4;
  int qt  = tmp % QT;   tmp /= QT;
  int head = tmp % NHc; int b = tmp / NHc;
  int q0 = qt * 16;

  int mrow   = q0 + (lane & 15);
  int mclamp = mrow < Qc ? mrow : (Qc - 1);
  int khalf  = (lane >> 4) * 16;
  const __bf16* ap = qproj + (((size_t)l * Bc + b) * Qc + mclamp) * HIDc
                           + head * HDc + khalf;
  v16bf a;
#pragma unroll
  for (int j = 0; j < 16; ++j)
    a[j] = ap[j];             // 32B contiguous

  size_t tile0 = ((size_t)(b * NHc + head) * PT + pt4 * NPT);
  float* orow = out + ((size_t)b * Qc * NHc + head) * (size_t)P;

#pragma unroll
  for (int u = 0; u < NPT; ++u) {
    int pt = pt4 * NPT + u;
    const __bf16* bp = kproj + (tile0 + u) * 512 + lane * 16;
    v16bf bm;
#pragma unroll
    for (int j = 0; j < 16; ++j) bm[j] = bp[j];       // 32B contiguous per lane

    v8f c = {};
    c = __builtin_amdgcn_wmma_f32_16x16x32_bf16(false, a, false, bm,
                                                (short)0, c, false, false);

    int  p      = pt * 16 + (lane & 15);
    bool masked = mask[(size_t)b * P + p] != 0;
#pragma unroll
    for (int r = 0; r < 8; ++r) {
      int m = q0 + r + ((lane >> 4) << 3);
      if (m < Qc) {
        float v = masked ? -__builtin_inff() : c[r];
        orow[(size_t)m * NHc * P + p] = v;
      }
    }
  }
}

// In-place softmax over the contiguous NH*P span for each (b,q).
// Pass 1: online (max,sum) (1 read); pass 2: normalize (1 read + 1 write).
__global__ void __launch_bounds__(256)
softmax_kernel(float* __restrict__ out, int L) {
  float* ptr = out + (size_t)blockIdx.x * L;
  int t = threadIdx.x;

  float m = -__builtin_inff(), s = 0.0f;
  for (int i = t; i < L; i += 256) {
    float x = ptr[i];
    if (x > m)      { s = s * __expf(m - x) + 1.0f; m = x; }
    else if (x != -__builtin_inff()) s += __expf(x - m);
  }

  __shared__ float sm[256], ss[256];
  sm[t] = m; ss[t] = s;
  __syncthreads();
  for (int off = 128; off > 0; off >>= 1) {
    if (t < off) {
      float m1 = sm[t], s1 = ss[t];
      float m2 = sm[t + off], s2 = ss[t + off];
      float nm = fmaxf(m1, m2);
      float e1 = (m1 == -__builtin_inff()) ? 0.0f : __expf(m1 - nm);
      float e2 = (m2 == -__builtin_inff()) ? 0.0f : __expf(m2 - nm);
      sm[t] = nm; ss[t] = s1 * e1 + s2 * e2;
    }
    __syncthreads();
  }
  float gm  = sm[0];
  float inv = 1.0f / ss[0];

  for (int i = t; i < L; i += 256) {
    float x = ptr[i];
    ptr[i] = __expf(x - gm) * inv;
  }
}

// ---------------------------------------------------------------------------
extern "C" void kernel_launch(void* const* d_in, const int* in_sizes, int n_in,
                              void* d_out, int out_size, void* d_ws, size_t ws_size,
                              hipStream_t stream) {
  (void)in_sizes; (void)n_in; (void)out_size; (void)ws_size;

  // setup_inputs() dict order: q, (k0,mask0)..(k3,mask3), w_q, b_q, w_k, b_k
  const float* q = (const float*)d_in[0];
  const float* kins[NLc]          = {(const float*)d_in[1], (const float*)d_in[3],
                                     (const float*)d_in[5], (const float*)d_in[7]};
  const unsigned char* masks[NLc] = {(const unsigned char*)d_in[2], (const unsigned char*)d_in[4],
                                     (const unsigned char*)d_in[6], (const unsigned char*)d_in[8]};
  const float* w_q = (const float*)d_in[9];
  const float* b_q = (const float*)d_in[10];
  const float* w_k = (const float*)d_in[11];
  const float* b_k = (const float*)d_in[12];
  float* out = (float*)d_out;

  // Workspace: qproj bf16 [NL,B,Q,HID] (1.2 MB), then swizzled kproj (~22 MB).
  __bf16* qproj = (__bf16*)d_ws;
  __bf16* kproj = qproj + (size_t)NLc * Bc * Qc * HIDc;

  const int Ps[NLc] = {128 * 128, 64 * 64, 32 * 32, 16 * 16};

  // Stage 1: q projections (all levels, one launch).
  {
    int waves  = NLc * Bc * QT * HT;
    int blocks = (waves * 32 + 255) / 256;
    proj_q_kernel<<<blocks, 256, 0, stream>>>(q, w_q, b_q, qproj);
  }

  // Stage 2: k projections per level (swizzled output, MT h-tiles per wave).
  {
    size_t koff = 0;
    for (int l = 0; l < NLc; ++l) {
      int P = Ps[l];
      int waves  = Bc * (HT / MT) * (P >> 4);
      int blocks = (waves * 32 + 255) / 256;
      proj_k_kernel<<<blocks, 256, 0, stream>>>(kins[l], w_k, b_k, kproj + koff, l, P);
      koff += (size_t)Bc * HIDc * P;
    }
  }

  // Stage 3 + 4: scores (WMMA) then in-place softmax, per level.
  {
    size_t koff = 0, ooff = 0;
    for (int l = 0; l < NLc; ++l) {
      int P = Ps[l];
      int waves  = Bc * NHc * QT * ((P >> 4) / NPT);
      int blocks = (waves * 32 + 255) / 256;
      attn_kernel<<<blocks, 256, 0, stream>>>(qproj, kproj + koff, masks[l],
                                              out + ooff, l, P);
      softmax_kernel<<<Bc * Qc, 256, 0, stream>>>(out + ooff, NHc * P);
      koff += (size_t)Bc * HIDc * P;
      ooff += (size_t)Bc * Qc * NHc * (size_t)P;
    }
  }
}
